// LSTMNetwork_50646254354817
// MI455X (gfx1250) — matrix operand order repaired and checked by reading, not compiled, not run on hardware
//
#include <hip/hip_runtime.h>
#include <hip/hip_bf16.h>
#include <stdint.h>

#define SEQ  8192
#define INP  1024
#define HID  2048
#define G4   (4 * HID)   // 8192 gate rows

// ---------------------------------------------------------------- WMMA types
typedef __bf16 v16bf __attribute__((ext_vector_type(16)));
typedef float  v8f   __attribute__((ext_vector_type(8)));

struct U32x8 { uint4 a, b; };   // 32-byte carrier for a v16bf fragment

__device__ __forceinline__ v16bf make_frag(uint4 lo, uint4 hi) {
    U32x8 t; t.a = lo; t.b = hi;
    return __builtin_bit_cast(v16bf, t);
}

// ---------------------------------------------------------------- helpers
__device__ __forceinline__ unsigned short f2bf_rne(float f) {
    unsigned u = __builtin_bit_cast(unsigned, f);
    unsigned r = (u + 0x7FFFu + ((u >> 16) & 1u)) >> 16;
    return (unsigned short)r;
}
__device__ __forceinline__ float bflo(unsigned u) {
    return __builtin_bit_cast(float, u << 16);
}
__device__ __forceinline__ float bfhi(unsigned u) {
    return __builtin_bit_cast(float, u & 0xFFFF0000u);
}
__device__ __forceinline__ float sigmoidf(float x) {
    return 1.0f / (1.0f + __expf(-x));
}
// 8-wide bf16 * fp32 dot: w holds 8 packed bf16, a/b hold 8 fp32 h values
__device__ __forceinline__ float dot8(uint4 w, float4 a, float4 b) {
    float s;
    s  = bflo(w.x) * a.x + bfhi(w.x) * a.y;
    s += bflo(w.y) * a.z + bfhi(w.y) * a.w;
    s += bflo(w.z) * b.x + bfhi(w.z) * b.y;
    s += bflo(w.w) * b.z + bfhi(w.w) * b.w;
    return s;
}

// ---------------------------------------------------------------- kernels
// fp32 -> bf16 (rne) bulk convert
__global__ void cvt_bf16_kernel(const float* __restrict__ in,
                                unsigned short* __restrict__ out, int n) {
    int i = blockIdx.x * blockDim.x + threadIdx.x;
    if (i < n) out[i] = f2bf_rne(in[i]);
}

// bias = b_ih + b_hh ; h = 0 ; bar = 0
__global__ void init_kernel(const float* __restrict__ b_ih,
                            const float* __restrict__ b_hh,
                            float* __restrict__ bias,
                            float* __restrict__ h,
                            unsigned* __restrict__ bar) {
    int i = blockIdx.x * blockDim.x + threadIdx.x;
    if (i < G4) bias[i] = b_ih[i] + b_hh[i];
    if (i < HID) h[i] = 0.0f;
    if (i == 0) *bar = 0u;
}

// x_proj[SEQ, G4] = A[SEQ, INP] @ W_ih[G4, INP]^T + bias   (bf16 WMMA, f32 acc)
// block = 256 threads = 8 waves; block tile M=128, N=64; each wave: 16x64
__global__ void gemm_xproj_kernel(const unsigned short* __restrict__ Abf,
                                  const unsigned short* __restrict__ Wbf,
                                  const float* __restrict__ bias,
                                  float* __restrict__ xproj) {
    const int wave = threadIdx.x >> 5;
    const int lane = threadIdx.x & 31;
    const int lr   = lane & 15;     // row / col within 16
    const int lh   = lane >> 4;     // lane-half selector
    const int m0   = blockIdx.x * 128 + wave * 16;
    const int n0   = blockIdx.y * 64;

    v8f acc0 = {}, acc1 = {}, acc2 = {}, acc3 = {};
    const unsigned short* arow = Abf + (size_t)(m0 + lr) * INP;

    for (int k = 0; k < INP; k += 32) {
        // A fragment (16x32 bf16): lane-half 0 -> K {k..k+15}
        //                          lane-half 1 -> K {k+16..k+31}
        const int ka = k + lh * 16;
        uint4 a_lo = *(const uint4*)(arow + ka);
        uint4 a_hi = *(const uint4*)(arow + ka + 8);
        v16bf av = make_frag(a_lo, a_hi);

        // B fragments: column n = W_ih row n; 16 contiguous K per lane-half
        const int kb = k + lh * 16;
#pragma unroll
        for (int nt = 0; nt < 4; ++nt) {
            const unsigned short* brow = Wbf + (size_t)(n0 + nt * 16 + lr) * INP;
            uint4 b_lo = *(const uint4*)(brow + kb);
            uint4 b_hi = *(const uint4*)(brow + kb + 8);
            v16bf bv = make_frag(b_lo, b_hi);
            switch (nt) {
            case 0: acc0 = __builtin_amdgcn_wmma_f32_16x16x32_bf16(false, av, false, bv, (short)0, acc0, false, false); break;
            case 1: acc1 = __builtin_amdgcn_wmma_f32_16x16x32_bf16(false, av, false, bv, (short)0, acc1, false, false); break;
            case 2: acc2 = __builtin_amdgcn_wmma_f32_16x16x32_bf16(false, av, false, bv, (short)0, acc2, false, false); break;
            case 3: acc3 = __builtin_amdgcn_wmma_f32_16x16x32_bf16(false, av, false, bv, (short)0, acc3, false, false); break;
            }
        }
    }

    // epilogue: C element (vgpr i, lane) -> M = i + lh*8, N = lr
    const int mrow = m0 + lh * 8;
#pragma unroll
    for (int nt = 0; nt < 4; ++nt) {
        const int ncol = n0 + nt * 16 + lr;
        const float bv = bias[ncol];
        const v8f acc = (nt == 0) ? acc0 : (nt == 1) ? acc1 : (nt == 2) ? acc2 : acc3;
#pragma unroll
        for (int i = 0; i < 8; ++i)
            xproj[(size_t)(mrow + i) * G4 + ncol] = acc[i] + bv;
    }
}

// Persistent LSTM scan. grid = 256 blocks x 256 threads (8 waves) = 2048 waves,
// wave w of block b owns hidden index j = b*8 + w for all 8192 timesteps.
// W_hh (bf16, 32 MB) stays L2-resident; one grid barrier per step.
__global__ void lstm_scan_kernel(const float* __restrict__ xproj,
                                 const unsigned short* __restrict__ Whh,  // [G4, HID] bf16
                                 float* __restrict__ h,
                                 float* __restrict__ hs,
                                 unsigned* __restrict__ bar) {
    const int wave = threadIdx.x >> 5;
    const int lane = threadIdx.x & 31;
    const int j    = blockIdx.x * 8 + wave;
    const unsigned nb = gridDim.x;

    const unsigned short* wi = Whh + (size_t)(j          ) * HID;
    const unsigned short* wf = Whh + (size_t)(j +     HID) * HID;
    const unsigned short* wg = Whh + (size_t)(j + 2 * HID) * HID;
    const unsigned short* wo = Whh + (size_t)(j + 3 * HID) * HID;

    float cj = 0.0f;   // cell state lives in a register for the whole scan

    for (int t = 0; t < SEQ; ++t) {
        float ai = 0.f, af = 0.f, ag = 0.f, ao = 0.f;
#pragma unroll
        for (int it = 0; it < HID / 256; ++it) {          // 8 iterations
            const int k = it * 256 + lane * 8;
            float4 h0 = *(const float4*)(h + k);
            float4 h1 = *(const float4*)(h + k + 4);
            uint4 w0 = *(const uint4*)(wi + k);
            uint4 w1 = *(const uint4*)(wf + k);
            uint4 w2 = *(const uint4*)(wg + k);
            uint4 w3 = *(const uint4*)(wo + k);
            ai += dot8(w0, h0, h1);
            af += dot8(w1, h0, h1);
            ag += dot8(w2, h0, h1);
            ao += dot8(w3, h0, h1);
        }
        // wave32 cross-lane reduce
#pragma unroll
        for (int off = 16; off; off >>= 1) {
            ai += __shfl_xor(ai, off, 32);
            af += __shfl_xor(af, off, 32);
            ag += __shfl_xor(ag, off, 32);
            ao += __shfl_xor(ao, off, 32);
        }
        if (lane == 0) {
            const float* xp = xproj + (size_t)t * G4;
            float gi = sigmoidf(xp[j          ] + ai);
            float gf = sigmoidf(xp[j +     HID] + af);
            float gg = tanhf   (xp[j + 2 * HID] + ag);
            float go = sigmoidf(xp[j + 3 * HID] + ao);
            cj = gf * cj + gi * gg;
            float hj = go * tanhf(cj);
            hs[(size_t)t * HID + j] = hj;
            h[j] = hj;
        }
        // ---- device-wide barrier (monotonic counter, release/acquire) ----
        __builtin_amdgcn_fence(__ATOMIC_RELEASE, "agent");
        __syncthreads();
        if (threadIdx.x == 0) {
            __hip_atomic_fetch_add(bar, 1u, __ATOMIC_RELAXED, __HIP_MEMORY_SCOPE_AGENT);
            const unsigned target = (unsigned)(t + 1) * nb;
            while (__hip_atomic_load(bar, __ATOMIC_ACQUIRE, __HIP_MEMORY_SCOPE_AGENT) < target)
                __builtin_amdgcn_s_sleep(1);
        }
        __syncthreads();
        __builtin_amdgcn_fence(__ATOMIC_ACQUIRE, "agent");
    }
}

// out[t] = sigmoid(dot(hs[t], W_lin) + b_lin); one wave per timestep
__global__ void final_linear_kernel(const float* __restrict__ hs,
                                    const float* __restrict__ Wlin,
                                    const float* __restrict__ blin,
                                    float* __restrict__ out) {
    const int wave = threadIdx.x >> 5;
    const int lane = threadIdx.x & 31;
    const int t    = blockIdx.x * 8 + wave;
    const float* hrow = hs + (size_t)t * HID;
    float a = 0.f;
#pragma unroll
    for (int it = 0; it < HID / 128; ++it) {
        const int k = it * 128 + lane * 4;
        float4 hv = *(const float4*)(hrow + k);
        float4 wv = *(const float4*)(Wlin + k);
        a += hv.x * wv.x + hv.y * wv.y + hv.z * wv.z + hv.w * wv.w;
    }
#pragma unroll
    for (int off = 16; off; off >>= 1) a += __shfl_xor(a, off, 32);
    if (lane == 0) out[t] = sigmoidf(a + blin[0]);
}

// ---------------------------------------------------------------- launcher
extern "C" void kernel_launch(void* const* d_in, const int* in_sizes, int n_in,
                              void* d_out, int out_size, void* d_ws, size_t ws_size,
                              hipStream_t stream) {
    (void)in_sizes; (void)n_in; (void)out_size; (void)ws_size;

    const float* input_seq = (const float*)d_in[0];   // [SEQ, INP]
    const float* W_ih      = (const float*)d_in[1];   // [G4, INP]
    const float* W_hh      = (const float*)d_in[2];   // [G4, HID]
    const float* b_ih      = (const float*)d_in[3];   // [G4]
    const float* b_hh      = (const float*)d_in[4];   // [G4]
    const float* W_lin     = (const float*)d_in[5];   // [1, HID]
    const float* b_lin     = (const float*)d_in[6];   // [1]
    float* out = (float*)d_out;                       // [SEQ]

    // workspace layout (256B-aligned chunks)
    char* ws = (char*)d_ws;
    float*          xproj  = (float*)(ws);                                   // 256 MB
    unsigned short* Abf    = (unsigned short*)(ws + (size_t)268435456);      // 16 MB
    unsigned short* Wihbf  = (unsigned short*)(ws + (size_t)285212672);      // 16 MB
    unsigned short* Whhbf  = (unsigned short*)(ws + (size_t)301989888);      // 32 MB
    float*          bias   = (float*)(ws + (size_t)335544320);               // 32 KB
    float*          hbuf   = (float*)(ws + (size_t)335577088);               // 8 KB
    float*          hs     = (float*)(ws + (size_t)335585280);               // 64 MB
    unsigned*       bar    = (unsigned*)(ws + (size_t)402694144);            // 4 B

    // 1) bf16 conversions
    {
        int n = SEQ * INP;
        cvt_bf16_kernel<<<(n + 255) / 256, 256, 0, stream>>>(input_seq, Abf, n);
        n = G4 * INP;
        cvt_bf16_kernel<<<(n + 255) / 256, 256, 0, stream>>>(W_ih, Wihbf, n);
        n = G4 * HID;
        cvt_bf16_kernel<<<(n + 255) / 256, 256, 0, stream>>>(W_hh, Whhbf, n);
    }
    // 2) bias fuse + state/barrier init (fresh every call -> deterministic)
    init_kernel<<<G4 / 256, 256, 0, stream>>>(b_ih, b_hh, bias, hbuf, bar);

    // 3) big input-projection GEMM on the WMMA pipe
    {
        dim3 grid(SEQ / 128, G4 / 64);
        gemm_xproj_kernel<<<grid, 256, 0, stream>>>(Abf, Wihbf, bias, xproj);
    }
    // 4) persistent sequential scan (2048 waves, one per hidden unit)
    lstm_scan_kernel<<<256, 256, 0, stream>>>(xproj, Whhbf, hbuf, hs, bar);

    // 5) readout
    final_linear_kernel<<<SEQ / 8, 256, 0, stream>>>(hs, W_lin, b_lin, out);
}
